// Encoder_42099269435400
// MI455X (gfx1250) — compile-verified, matrix operand order checked by reading, not run on hardware
//
#include <hip/hip_runtime.h>

typedef __bf16 bf16_t;
typedef __attribute__((ext_vector_type(16))) __bf16 v16bf;
typedef __attribute__((ext_vector_type(8)))  float  v8f;

#define D_MODEL 1024
#define SEQ     1024
#define BATCHN  4
#define NLAYERS 4
#define MROWS   4096   // BATCH*SEQ

// K-pair offset pattern for 16-bit A/B WMMA fragments (ISA 7.12.2):
// lane-half h, vgpr v -> first K index of the packed pair
__device__ __forceinline__ int ko_idx(int v, int h) {
  return ((v >> 2) << 4) | ((v & 3) << 1) | (h << 3);
}

union FragU { unsigned int u[8]; v16bf v; };
union PackU { unsigned int u; bf16_t h[2]; };

__device__ __forceinline__ v8f wmma_bf16(v16bf a, v16bf b, v8f c) {
  return __builtin_amdgcn_wmma_f32_16x16x32_bf16(false, a, false, b, (short)0, c,
                                                 false, false);
}

// ---- DPP16 xor-butterfly reductions within a 16-lane row (VALU, no LDS) ----
// 0xB1 = quad_perm(1,0,3,2)  (xor 1)
// 0x4E = quad_perm(2,3,0,1)  (xor 2)
// 0x141 = ROW_HALF_MIRROR    (i ^ 7 within 8)
// 0x140 = ROW_MIRROR         (i ^ 15 within 16)
template <int CTRL>
__device__ __forceinline__ float dpp_mov(float x) {
  return __builtin_bit_cast(
      float, __builtin_amdgcn_update_dpp(0, __builtin_bit_cast(int, x), CTRL,
                                         0xF, 0xF, true));
}
__device__ __forceinline__ float row16_max(float x) {
  x = fmaxf(x, dpp_mov<0xB1>(x));
  x = fmaxf(x, dpp_mov<0x4E>(x));
  x = fmaxf(x, dpp_mov<0x141>(x));
  x = fmaxf(x, dpp_mov<0x140>(x));
  return x;
}
__device__ __forceinline__ float row16_sum(float x) {
  x += dpp_mov<0xB1>(x);
  x += dpp_mov<0x4E>(x);
  x += dpp_mov<0x141>(x);
  x += dpp_mov<0x140>(x);
  return x;
}

// ------------------------------------------------- fp32 -> bf16 (paired store)
__global__ __launch_bounds__(256) void cvt_kernel(const float* __restrict__ in,
                                                  unsigned int* __restrict__ out,
                                                  int npairs) {
  int i = blockIdx.x * 256 + threadIdx.x;
  int stride = gridDim.x * 256;
  const float2* in2 = (const float2*)in;
  for (; i < npairs; i += stride) {
    float2 a = in2[i];
    PackU p;
    p.h[0] = (bf16_t)a.x;
    p.h[1] = (bf16_t)a.y;
    out[i] = p.u;
  }
}

// ------------------------------------------------- embedding + position encode
__global__ __launch_bounds__(256) void embed_kernel(const int* __restrict__ x,
                                                    const float* __restrict__ emb,
                                                    float* __restrict__ h,
                                                    bf16_t* __restrict__ hbf) {
  const int row = blockIdx.x;            // b*SEQ + s
  const int s = row & (SEQ - 1);
  const int tok = x[row];
  const float pos = (float)s;
  const int d0 = threadIdx.x * 4;
  float val[4];
#pragma unroll
  for (int i = 0; i < 4; ++i) {
    int d = d0 + i;
    float pe;
    if (d < D_MODEL / 2) {
      float ang = pos * __powf(10000.0f, (float)(2 * d) * (1.0f / (float)D_MODEL));
      pe = __sinf(ang);
    } else {
      int j = d - D_MODEL / 2;
      float ang = pos * __powf(10000.0f, (float)(2 * j) * (1.0f / (float)D_MODEL));
      pe = __cosf(ang);
    }
    val[i] = emb[(size_t)tok * D_MODEL + d] * 32.0f + pe;  // sqrt(1024)=32
  }
  size_t idx0 = (size_t)row * D_MODEL + d0;
  *(float4*)&h[idx0] = make_float4(val[0], val[1], val[2], val[3]);
  PackU p0, p1;
  p0.h[0] = (bf16_t)val[0]; p0.h[1] = (bf16_t)val[1];
  p1.h[0] = (bf16_t)val[2]; p1.h[1] = (bf16_t)val[3];
  *(unsigned int*)&hbf[idx0]     = p0.u;
  *(unsigned int*)&hbf[idx0 + 2] = p1.u;
}

// ------------------------------------------------------------------- GEMM bf16
// C[4096,1024] = A[4096,1024](bf16) @ B[1024,1024](bf16) + bias
// block: 256 thr (8 waves), tile 128x128, K-slab 64; wave tile 32x64
template <bool OUT_BF16>
__global__ __launch_bounds__(256) void gemm_kernel(const bf16_t* __restrict__ A,
                                                   const bf16_t* __restrict__ B,
                                                   const float* __restrict__ bias,
                                                   void* __restrict__ out) {
  constexpr int N = D_MODEL, K = D_MODEL;
  __shared__ bf16_t As[128][66];   // [m][k], +2 pad
  __shared__ bf16_t Bs[128][66];   // transposed: [n][k], +2 pad

  const int tid = threadIdx.x;
  const int mBase = blockIdx.x * 128;
  const int nBase = blockIdx.y * 128;
  const int wave = tid >> 5, lane = tid & 31;
  const int r = lane & 15, hh = lane >> 4;
  const int wm = wave & 3, wn = wave >> 2;

  v8f acc[2][4];
#pragma unroll
  for (int mi = 0; mi < 2; ++mi)
#pragma unroll
    for (int ni = 0; ni < 4; ++ni)
#pragma unroll
      for (int e = 0; e < 8; ++e) acc[mi][ni][e] = 0.0f;

  for (int kb = 0; kb < K; kb += 64) {
    __syncthreads();
    // stage A slab 128x64 (row major)
    for (int j = tid; j < 128 * 32; j += 256) {
      int row = j >> 5, c2 = (j & 31) << 1;
      *(unsigned int*)&As[row][c2] =
          *(const unsigned int*)(A + (size_t)(mBase + row) * K + kb + c2);
    }
    // stage B slab 64x128 transposed into [n][k]
    for (int j = tid; j < 64 * 64; j += 256) {
      int k = j >> 6, n = (j & 63) << 1;
      PackU p;
      p.u = *(const unsigned int*)(B + (size_t)(kb + k) * N + nBase + n);
      Bs[n][k] = p.h[0];
      Bs[n + 1][k] = p.h[1];
    }
    // prefetch next slabs while this one computes
    if (kb + 64 < K) {
      {
        int row = tid >> 1, half = tid & 1;  // 128 rows x 2 x 64B
        __builtin_prefetch(A + (size_t)(mBase + row) * K + kb + 64 + half * 32, 0, 0);
      }
      {
        int row = tid >> 2, q = tid & 3;     // 64 rows x 4 x 64B
        __builtin_prefetch(B + (size_t)(kb + 64 + row) * N + nBase + q * 32, 0, 0);
      }
    }
    __syncthreads();

#pragma unroll
    for (int kk = 0; kk < 2; ++kk) {
      v16bf afr[2], bfr[4];
#pragma unroll
      for (int mi = 0; mi < 2; ++mi) {
        FragU f;
        int row = wm * 32 + mi * 16 + r;
#pragma unroll
        for (int v = 0; v < 8; ++v)
          f.u[v] = *(const unsigned int*)&As[row][kk * 32 + ko_idx(v, hh)];
        afr[mi] = f.v;
      }
#pragma unroll
      for (int ni = 0; ni < 4; ++ni) {
        FragU f;
        int col = wn * 64 + ni * 16 + r;
#pragma unroll
        for (int v = 0; v < 8; ++v)
          f.u[v] = *(const unsigned int*)&Bs[col][kk * 32 + ko_idx(v, hh)];
        bfr[ni] = f.v;
      }
#pragma unroll
      for (int mi = 0; mi < 2; ++mi)
#pragma unroll
        for (int ni = 0; ni < 4; ++ni)
          acc[mi][ni] = wmma_bf16(afr[mi], bfr[ni], acc[mi][ni]);
    }
  }

#pragma unroll
  for (int mi = 0; mi < 2; ++mi)
#pragma unroll
    for (int ni = 0; ni < 4; ++ni) {
      int ncol = nBase + wn * 64 + ni * 16 + r;
      float bv = bias[ncol];
#pragma unroll
      for (int v = 0; v < 8; ++v) {
        int mrow = mBase + wm * 32 + mi * 16 + v + 8 * hh;
        float val = acc[mi][ni][v] + bv;
        size_t idx = (size_t)mrow * N + ncol;
        if (OUT_BF16) ((bf16_t*)out)[idx] = (bf16_t)val;
        else          ((float*)out)[idx]  = val;
      }
    }
}

// -------------------------------------------------------- flash attention bf16
// grid: (SEQ/128, NHEADS, BATCH); block 256 (8 waves, 16 q-rows each)
__global__ __launch_bounds__(256) void attn_kernel(const bf16_t* __restrict__ Q,
                                                   const bf16_t* __restrict__ Kp,
                                                   const bf16_t* __restrict__ V,
                                                   const float* __restrict__ mask,
                                                   bf16_t* __restrict__ ctxout) {
  constexpr int S = SEQ, D = D_MODEL;
  __shared__ bf16_t Qs[128][68];     // q rows x dh
  __shared__ bf16_t Ks[32][68];      // kpos x dh  (== B^T layout for Q@K^T)
  __shared__ bf16_t Vs[64][36];      // dh x kpos  (B^T layout for P@V)
  __shared__ float  Pb[8][16][33];   // per-wave P tile 16x32

  const int tid = threadIdx.x, wave = tid >> 5, lane = tid & 31;
  const int r = lane & 15, hh = lane >> 4;
  const int qb = blockIdx.x * 128;
  const int hbase = blockIdx.y * 64;
  const int bz = blockIdx.z;
  const size_t rowOff = (size_t)bz * S;

  for (int j = tid; j < 128 * 32; j += 256) {
    int row = j >> 5, c2 = (j & 31) << 1;
    *(unsigned int*)&Qs[row][c2] =
        *(const unsigned int*)(Q + (rowOff + qb + row) * D + hbase + c2);
  }
  __syncthreads();

  v16bf aq[2];
#pragma unroll
  for (int kk = 0; kk < 2; ++kk) {
    FragU f;
    int row = wave * 16 + r;
#pragma unroll
    for (int v = 0; v < 8; ++v)
      f.u[v] = *(const unsigned int*)&Qs[row][kk * 32 + ko_idx(v, hh)];
    aq[kk] = f.v;
  }

  v8f ctx[4];
  float mrun[8], lrun[8];
#pragma unroll
  for (int dt = 0; dt < 4; ++dt)
#pragma unroll
    for (int v = 0; v < 8; ++v) ctx[dt][v] = 0.0f;
#pragma unroll
  for (int v = 0; v < 8; ++v) { mrun[v] = -1.0e30f; lrun[v] = 0.0f; }

  for (int it = 0; it < 32; ++it) {
    __syncthreads();
    for (int j = tid; j < 32 * 32; j += 256) {
      int kr = j >> 5, c2 = (j & 31) << 1;
      *(unsigned int*)&Ks[kr][c2] =
          *(const unsigned int*)(Kp + (rowOff + it * 32 + kr) * D + hbase + c2);
    }
    for (int j = tid; j < 32 * 32; j += 256) {
      int kr = j >> 5, c2 = (j & 31) << 1;
      PackU p;
      p.u = *(const unsigned int*)(V + (rowOff + it * 32 + kr) * D + hbase + c2);
      Vs[c2][kr] = p.h[0];
      Vs[c2 + 1][kr] = p.h[1];
    }
    // prefetch next K/V tiles (lowers to global_prefetch_b8)
    if (it + 1 < 32) {
      if (tid < 64) {
        int kr = tid >> 1, half = tid & 1;
        __builtin_prefetch(Kp + (rowOff + (it + 1) * 32 + kr) * D + hbase + half * 32, 0, 0);
      } else if (tid < 128) {
        int t2 = tid - 64;
        int kr = t2 >> 1, half = t2 & 1;
        __builtin_prefetch(V + (rowOff + (it + 1) * 32 + kr) * D + hbase + half * 32, 0, 0);
      }
    }
    __syncthreads();

    // scores: two 16-wide kpos tiles, each A(16x64 over dh) @ K^T
    float s0[8], s1[8];
    {
      v8f c;
#pragma unroll
      for (int v = 0; v < 8; ++v) c[v] = 0.0f;
#pragma unroll
      for (int kk = 0; kk < 2; ++kk) {
        FragU f;
#pragma unroll
        for (int v = 0; v < 8; ++v)
          f.u[v] = *(const unsigned int*)&Ks[r][kk * 32 + ko_idx(v, hh)];
        c = wmma_bf16(aq[kk], f.v, c);
      }
      float mv = mask[rowOff + it * 32 + r] * -1.0e9f;
#pragma unroll
      for (int v = 0; v < 8; ++v) s0[v] = c[v] * 0.125f + mv;
    }
    {
      v8f c;
#pragma unroll
      for (int v = 0; v < 8; ++v) c[v] = 0.0f;
#pragma unroll
      for (int kk = 0; kk < 2; ++kk) {
        FragU f;
#pragma unroll
        for (int v = 0; v < 8; ++v)
          f.u[v] = *(const unsigned int*)&Ks[16 + r][kk * 32 + ko_idx(v, hh)];
        c = wmma_bf16(aq[kk], f.v, c);
      }
      float mv = mask[rowOff + it * 32 + 16 + r] * -1.0e9f;
#pragma unroll
      for (int v = 0; v < 8; ++v) s1[v] = c[v] * 0.125f + mv;
    }

    // online softmax: rows live at VGPR index v (row = v + 8*hh);
    // N-dim reduction across the 16-lane row via DPP16 xor butterfly (pure VALU)
#pragma unroll
    for (int v = 0; v < 8; ++v) {
      float tmax = row16_max(fmaxf(s0[v], s1[v]));
      float mn = fmaxf(mrun[v], tmax);
      float corr = __expf(mrun[v] - mn);
      float p0 = __expf(s0[v] - mn);
      float p1 = __expf(s1[v] - mn);
      float rs = row16_sum(p0 + p1);
      lrun[v] = lrun[v] * corr + rs;
      mrun[v] = mn;
#pragma unroll
      for (int dt = 0; dt < 4; ++dt) ctx[dt][v] *= corr;
      Pb[wave][v + 8 * hh][r] = p0;
      Pb[wave][v + 8 * hh][16 + r] = p1;
    }

    // redistribute P (C layout -> A layout) through wave-private LDS
    FragU ap;
#pragma unroll
    for (int v = 0; v < 8; ++v) {
      int k = ko_idx(v, hh);
      PackU p;
      p.h[0] = (bf16_t)Pb[wave][r][k];
      p.h[1] = (bf16_t)Pb[wave][r][k + 1];
      ap.u[v] = p.u;
    }
#pragma unroll
    for (int dt = 0; dt < 4; ++dt) {
      FragU fb;
#pragma unroll
      for (int v = 0; v < 8; ++v)
        fb.u[v] = *(const unsigned int*)&Vs[dt * 16 + r][ko_idx(v, hh)];
      ctx[dt] = wmma_bf16(ap.v, fb.v, ctx[dt]);
    }
  }

#pragma unroll
  for (int dt = 0; dt < 4; ++dt)
#pragma unroll
    for (int v = 0; v < 8; ++v) {
      float val = ctx[dt][v] / lrun[v];
      int qrow = qb + wave * 16 + v + 8 * hh;
      ctxout[(rowOff + qrow) * D + hbase + dt * 16 + r] = (bf16_t)val;
    }
}

// ----------------------------------------------------- residual add + layernorm
__global__ __launch_bounds__(256) void add_ln_kernel(float* __restrict__ h,
                                                     const float* __restrict__ dlt,
                                                     const float* __restrict__ g,
                                                     const float* __restrict__ b,
                                                     bf16_t* __restrict__ hbf) {
  __shared__ float s1[256], s2[256];
  const int row = blockIdx.x, t = threadIdx.x;
  const int d0 = t * 4;
  const size_t idx0 = (size_t)row * D_MODEL + d0;
  float4 hv = *(const float4*)&h[idx0];
  float4 dv = *(const float4*)&dlt[idx0];
  float x[4] = {hv.x + dv.x, hv.y + dv.y, hv.z + dv.z, hv.w + dv.w};
  float sum = x[0] + x[1] + x[2] + x[3];
  float ss = x[0] * x[0] + x[1] * x[1] + x[2] * x[2] + x[3] * x[3];
  s1[t] = sum; s2[t] = ss;
  __syncthreads();
  for (int o = 128; o > 0; o >>= 1) {
    if (t < o) { s1[t] += s1[t + o]; s2[t] += s2[t + o]; }
    __syncthreads();
  }
  float mean = s1[0] * (1.0f / (float)D_MODEL);
  float var = s2[0] * (1.0f / (float)D_MODEL) - mean * mean;
  float rstd = rsqrtf(var + 1e-6f);
  float4 gv = *(const float4*)&g[d0];
  float4 bv = *(const float4*)&b[d0];
  float y[4];
  y[0] = (x[0] - mean) * rstd * gv.x + bv.x;
  y[1] = (x[1] - mean) * rstd * gv.y + bv.y;
  y[2] = (x[2] - mean) * rstd * gv.z + bv.z;
  y[3] = (x[3] - mean) * rstd * gv.w + bv.w;
  *(float4*)&h[idx0] = make_float4(y[0], y[1], y[2], y[3]);
  PackU p0, p1;
  p0.h[0] = (bf16_t)y[0]; p0.h[1] = (bf16_t)y[1];
  p1.h[0] = (bf16_t)y[2]; p1.h[1] = (bf16_t)y[3];
  *(unsigned int*)&hbf[idx0]     = p0.u;
  *(unsigned int*)&hbf[idx0 + 2] = p1.u;
}

// ------------------------------------------------------------------------ host
extern "C" void kernel_launch(void* const* d_in, const int* in_sizes, int n_in,
                              void* d_out, int out_size, void* d_ws, size_t ws_size,
                              hipStream_t stream) {
  (void)in_sizes; (void)n_in; (void)out_size; (void)ws_size;
  const int*   x    = (const int*)d_in[0];
  const float* mask = (const float*)d_in[1];
  const float* emb  = (const float*)d_in[2];
  const float* wq   = (const float*)d_in[3];
  const float* bq   = (const float*)d_in[4];
  const float* wk   = (const float*)d_in[5];
  const float* bk   = (const float*)d_in[6];
  const float* wv   = (const float*)d_in[7];
  const float* bv   = (const float*)d_in[8];
  const float* wo   = (const float*)d_in[9];
  const float* bo   = (const float*)d_in[10];
  const float* wf   = (const float*)d_in[11];
  const float* bff  = (const float*)d_in[12];
  const float* g1   = (const float*)d_in[13];
  const float* b1   = (const float*)d_in[14];
  const float* g2   = (const float*)d_in[15];
  const float* b2   = (const float*)d_in[16];

  float* h = (float*)d_out;
  char* ws = (char*)d_ws;
  const size_t MB = 1u << 20;
  bf16_t* h_bf  = (bf16_t*)(ws);
  bf16_t* qbuf  = (bf16_t*)(ws + 8 * MB);
  bf16_t* kbuf  = (bf16_t*)(ws + 16 * MB);
  bf16_t* vbuf  = (bf16_t*)(ws + 24 * MB);
  bf16_t* cbuf  = (bf16_t*)(ws + 32 * MB);
  float*  tmp   = (float*)(ws + 40 * MB);
  bf16_t* wqb   = (bf16_t*)(ws + 56 * MB);
  bf16_t* wkb   = (bf16_t*)(ws + 64 * MB);
  bf16_t* wvb   = (bf16_t*)(ws + 72 * MB);
  bf16_t* wob   = (bf16_t*)(ws + 80 * MB);
  bf16_t* wfb   = (bf16_t*)(ws + 88 * MB);

  const int NP = NLAYERS * D_MODEL * D_MODEL / 2;  // pairs
  cvt_kernel<<<2048, 256, 0, stream>>>(wq, (unsigned int*)wqb, NP);
  cvt_kernel<<<2048, 256, 0, stream>>>(wk, (unsigned int*)wkb, NP);
  cvt_kernel<<<2048, 256, 0, stream>>>(wv, (unsigned int*)wvb, NP);
  cvt_kernel<<<2048, 256, 0, stream>>>(wo, (unsigned int*)wob, NP);
  cvt_kernel<<<2048, 256, 0, stream>>>(wf, (unsigned int*)wfb, NP);

  embed_kernel<<<MROWS, 256, 0, stream>>>(x, emb, h, h_bf);

  const dim3 ggrid(MROWS / 128, D_MODEL / 128);
  const dim3 agrid(SEQ / 128, 16, BATCHN);
  for (int l = 0; l < NLAYERS; ++l) {
    const size_t ow = (size_t)l * D_MODEL * D_MODEL;
    const size_t ob = (size_t)l * D_MODEL;
    gemm_kernel<true><<<ggrid, 256, 0, stream>>>(h_bf, wqb + ow, bq + ob, qbuf);
    gemm_kernel<true><<<ggrid, 256, 0, stream>>>(h_bf, wkb + ow, bk + ob, kbuf);
    gemm_kernel<true><<<ggrid, 256, 0, stream>>>(h_bf, wvb + ow, bv + ob, vbuf);
    attn_kernel<<<agrid, 256, 0, stream>>>(qbuf, kbuf, vbuf, mask, cbuf);
    gemm_kernel<false><<<ggrid, 256, 0, stream>>>(cbuf, wob + ow, bo + ob, tmp);
    add_ln_kernel<<<MROWS, 256, 0, stream>>>(h, tmp, g1 + ob, b1 + ob, h_bf);
    gemm_kernel<false><<<ggrid, 256, 0, stream>>>(h_bf, wfb + ow, bff + ob, tmp);
    add_ln_kernel<<<MROWS, 256, 0, stream>>>(h, tmp, g2 + ob, b2 + ob, h_bf);
  }
}
